// _FusedQKVAttentionWrapper_12051678232737
// MI455X (gfx1250) — compile-verified, matrix operand order checked by reading, not run on hardware
//
#include <hip/hip_runtime.h>
#include <hip/hip_bf16.h>

#define T_SEQ   2048
#define D_MODEL 4096
#define NH      32
#define NKV     8
#define HD      128
#define QSZ     (NH * HD)           // 4096
#define KSZ     (NKV * HD)          // 1024
#define QKV_N   (QSZ + 2 * KSZ)     // 6144
#define BATCH   2
#define ROWS    (BATCH * T_SEQ)     // 4096
#define SOFT_SCALE 0.08838834764831845f   // 1/sqrt(128), folded into Q

typedef __bf16 bf16_t;
typedef bf16_t v16bf __attribute__((ext_vector_type(16)));
typedef bf16_t v4bf  __attribute__((ext_vector_type(4)));
typedef float  v8f   __attribute__((ext_vector_type(8)));
typedef int    v4i   __attribute__((ext_vector_type(4)));

__device__ __forceinline__ v8f wmma_bf16(v16bf a, v16bf b, v8f c) {
  // D = A(16x32 bf16) * B(32x16 bf16) + C(16x16 f32)
  return __builtin_amdgcn_wmma_f32_16x16x32_bf16(
      /*neg_a=*/false, a, /*neg_b=*/false, b,
      /*c_mod=*/(short)0, c, /*reuse_a=*/false, /*reuse_b=*/false);
}

// ---- CDNA5 async global->LDS copy (ASYNCcnt-tracked), 16B per lane --------
__device__ __forceinline__ void async_copy16(bf16_t* lds_dst, const bf16_t* g_src) {
#if __has_builtin(__builtin_amdgcn_global_load_async_to_lds_b128)
  __builtin_amdgcn_global_load_async_to_lds_b128(
      (__attribute__((address_space(1))) v4i*)g_src,
      (__attribute__((address_space(3))) v4i*)lds_dst, 0, 0);
#else
  unsigned off = (unsigned)(uintptr_t)lds_dst;        // flat low 32 = LDS offset
  unsigned long long ga = (unsigned long long)(uintptr_t)g_src;
  asm volatile("global_load_async_to_lds_b128 %0, %1, off"
               :: "v"(off), "v"(ga) : "memory");
#endif
}
__device__ __forceinline__ void wait_async0() {
#if __has_builtin(__builtin_amdgcn_s_wait_asynccnt)
  __builtin_amdgcn_s_wait_asynccnt(0);
#else
  asm volatile("s_wait_asynccnt 0" ::: "memory");
#endif
}
__device__ __forceinline__ void wait_async4() {
#if __has_builtin(__builtin_amdgcn_s_wait_asynccnt)
  __builtin_amdgcn_s_wait_asynccnt(4);
#else
  asm volatile("s_wait_asynccnt 4" ::: "memory");
#endif
}

// ---------------------------------------------------------------------------
// fp32 -> bf16 bulk convert (one-time, bandwidth-bound prep pass)
// ---------------------------------------------------------------------------
__global__ __launch_bounds__(256)
void cvt_f32_bf16_kernel(const float* __restrict__ src, bf16_t* __restrict__ dst,
                         int n4) {
  int i = blockIdx.x * 256 + threadIdx.x;
  int stride = gridDim.x * 256;
  for (; i < n4; i += stride) {
    float4 v = reinterpret_cast<const float4*>(src)[i];
    v4bf o;
    o.x = (bf16_t)v.x; o.y = (bf16_t)v.y; o.z = (bf16_t)v.z; o.w = (bf16_t)v.w;
    reinterpret_cast<v4bf*>(dst)[i] = o;
  }
}

// ---------------------------------------------------------------------------
// Tiled WMMA GEMM:  Out[M,N] = A[M,K] * W[N,K]^T  (A, W bf16)
// Block tile 128x128, 8 waves (4Mx2N), 2x4 16x16 accum per wave.
// K stepped by 32; tiles streamed with double-buffered async-to-LDS copies:
// each thread owns four 16B chunks per stage (2 for A, 2 for W).
// ---------------------------------------------------------------------------
template <bool OUT_F32>
__global__ __launch_bounds__(256)
void gemm_wmma_async(const bf16_t* __restrict__ A, const bf16_t* __restrict__ W,
                     void* __restrict__ Outp, int M, int N, int K) {
  __shared__ __align__(32) bf16_t sA[2][128 * 32];
  __shared__ __align__(32) bf16_t sB[2][128 * 32];

  const int tid   = threadIdx.x;
  const int lane  = tid & 31;
  const int wave  = tid >> 5;
  const int waveM = wave >> 1;          // 0..3 -> 32 rows each
  const int waveN = wave & 1;           // 0..1 -> 64 cols each
  const int lrow  = lane & 15;
  const int hi    = lane >> 4;
  const int ksel  = hi * 16;

  const int row0 = blockIdx.y * 128;
  const int col0 = blockIdx.x * 128;

  // staging ownership: rows r0 and r0+64, 8-element (16B) k-chunk k8
  const int r0 = tid >> 2;              // 0..63
  const int k8 = (tid & 3) * 8;         // 0,8,16,24

  const bf16_t* gA0 = A + (size_t)(row0 + r0) * K + k8;
  const bf16_t* gA1 = A + (size_t)(row0 + r0 + 64) * K + k8;
  const bf16_t* gB0 = W + (size_t)(col0 + r0) * K + k8;
  const bf16_t* gB1 = W + (size_t)(col0 + r0 + 64) * K + k8;

  bf16_t* dA0[2], *dA1[2], *dB0[2], *dB1[2];
#pragma unroll
  for (int bidx = 0; bidx < 2; ++bidx) {
    dA0[bidx] = &sA[bidx][r0 * 32 + k8];
    dA1[bidx] = &sA[bidx][(r0 + 64) * 32 + k8];
    dB0[bidx] = &sB[bidx][r0 * 32 + k8];
    dB1[bidx] = &sB[bidx][(r0 + 64) * 32 + k8];
  }

  // prologue: stage 0 into buffer 0
  async_copy16(dA0[0], gA0);
  async_copy16(dA1[0], gA1);
  async_copy16(dB0[0], gB0);
  async_copy16(dB1[0], gB1);

  const int nsteps = K / 32;
  v8f acc[2][4] = {};

  for (int step = 0; step < nsteps; ++step) {
    const int cur = step & 1;
    if (step + 1 < nsteps) {
      const int k0n = (step + 1) * 32;
      async_copy16(dA0[cur ^ 1], gA0 + k0n);
      async_copy16(dA1[cur ^ 1], gA1 + k0n);
      async_copy16(dB0[cur ^ 1], gB0 + k0n);
      async_copy16(dB1[cur ^ 1], gB1 + k0n);
      wait_async4();        // async loads retire in order: stage `step` done
    } else {
      wait_async0();
    }
    __syncthreads();

    v16bf af[2], bfr[4];
#pragma unroll
    for (int mt = 0; mt < 2; ++mt)
      af[mt] = *reinterpret_cast<const v16bf*>(
          &sA[cur][(waveM * 32 + mt * 16 + lrow) * 32 + ksel]);
#pragma unroll
    for (int nt = 0; nt < 4; ++nt)
      bfr[nt] = *reinterpret_cast<const v16bf*>(
          &sB[cur][(waveN * 64 + nt * 16 + lrow) * 32 + ksel]);
#pragma unroll
    for (int mt = 0; mt < 2; ++mt)
#pragma unroll
      for (int nt = 0; nt < 4; ++nt)
        acc[mt][nt] = wmma_bf16(af[mt], bfr[nt], acc[mt][nt]);

    __syncthreads();        // everyone done reading buf `cur` before reuse
  }

  // epilogue: C layout -> VGPR r holds (row r / r+8) across 16-lane halves
#pragma unroll
  for (int mt = 0; mt < 2; ++mt) {
#pragma unroll
    for (int nt = 0; nt < 4; ++nt) {
#pragma unroll
      for (int r = 0; r < 8; ++r) {
        int row = row0 + waveM * 32 + mt * 16 + r + 8 * hi;
        int col = col0 + waveN * 64 + nt * 16 + lrow;
        float v = acc[mt][nt][r];
        if constexpr (OUT_F32)
          ((float*)Outp)[(size_t)row * N + col] = v;
        else
          ((bf16_t*)Outp)[(size_t)row * N + col] = (bf16_t)v;
      }
    }
  }
}

// ---------------------------------------------------------------------------
// RoPE + layout shuffle.  Reads bf16 qkv [ROWS, 6144]; writes
//   Q [B,NH,T,HD]  (rope applied, pre-scaled by 1/sqrt(HD))
//   K [B,NKV,T,HD] (rope applied)
//   Vt[B,NKV,HD,T] (transposed: PV WMMA B-fragment becomes contiguous)
// ---------------------------------------------------------------------------
__global__ __launch_bounds__(256)
void rope_kernel(const bf16_t* __restrict__ qkv,
                 const float* __restrict__ fcos, const float* __restrict__ fsin,
                 bf16_t* __restrict__ Qb, bf16_t* __restrict__ Kb,
                 bf16_t* __restrict__ Vt) {
  const int row = blockIdx.x;                 // b*T + t
  const int b = row / T_SEQ;
  const int t = row - b * T_SEQ;
  const bf16_t* src = qkv + (size_t)row * QKV_N;
  const float* cr = fcos + t * (HD / 2);
  const float* sr = fsin + t * (HD / 2);

  for (int p = threadIdx.x; p < NH * (HD / 2); p += 256) {
    int h = p >> 6, d2 = p & 63;
    float x0 = (float)src[h * HD + 2 * d2];
    float x1 = (float)src[h * HD + 2 * d2 + 1];
    float c = cr[d2], s = sr[d2];
    size_t o = ((size_t)(b * NH + h) * T_SEQ + t) * HD + 2 * d2;
    Qb[o]     = (bf16_t)((x0 * c - x1 * s) * SOFT_SCALE);
    Qb[o + 1] = (bf16_t)((x0 * s + x1 * c) * SOFT_SCALE);
  }
  for (int p = threadIdx.x; p < NKV * (HD / 2); p += 256) {
    int h = p >> 6, d2 = p & 63;
    float x0 = (float)src[QSZ + h * HD + 2 * d2];
    float x1 = (float)src[QSZ + h * HD + 2 * d2 + 1];
    float c = cr[d2], s = sr[d2];
    size_t o = ((size_t)(b * NKV + h) * T_SEQ + t) * HD + 2 * d2;
    Kb[o]     = (bf16_t)(x0 * c - x1 * s);
    Kb[o + 1] = (bf16_t)(x0 * s + x1 * c);
  }
  for (int e = threadIdx.x; e < KSZ; e += 256) {
    int h = e >> 7, d = e & 127;
    Vt[((size_t)(b * NKV + h) * HD + d) * T_SEQ + t] = src[QSZ + KSZ + e];
  }
}

// ---------------------------------------------------------------------------
// Causal flash attention. One 16-query tile per wave; key tiles of 32;
// online softmax with 16-lane shfl reductions (C-frag row layout);
// P re-laid C-frag -> A-frag through per-wave LDS. No block barriers.
// ---------------------------------------------------------------------------
__global__ __launch_bounds__(256)
void attn_kernel(const bf16_t* __restrict__ Qb, const bf16_t* __restrict__ Kb,
                 const bf16_t* __restrict__ Vt, bf16_t* __restrict__ Y) {
  __shared__ __align__(32) bf16_t sP[8][16][32];

  const int lane = threadIdx.x & 31;
  const int wave = threadIdx.x >> 5;
  const int lrow = lane & 15;
  const int hi   = lane >> 4;
  const int h    = blockIdx.y;
  const int b    = blockIdx.z;
  const int hkv  = h >> 2;                       // GQA: 4 q-heads per kv-head
  const int qb0  = blockIdx.x * 128 + wave * 16;

  const bf16_t* Qp = Qb + (size_t)(b * NH + h) * T_SEQ * HD;
  const bf16_t* Kp = Kb + (size_t)(b * NKV + hkv) * T_SEQ * HD;
  const bf16_t* Vp = Vt + (size_t)(b * NKV + hkv) * HD * T_SEQ;

  v16bf qf[4];
#pragma unroll
  for (int c = 0; c < 4; ++c)
    qf[c] = *reinterpret_cast<const v16bf*>(
        &Qp[(size_t)(qb0 + lrow) * HD + c * 32 + hi * 16]);

  v8f o[8] = {};
  float m[8], l[8];
#pragma unroll
  for (int r = 0; r < 8; ++r) { m[r] = -1e30f; l[r] = 0.f; }

  for (int kb = 0; kb < qb0 + 16; kb += 32) {
    v8f s0 = {}, s1 = {};
#pragma unroll
    for (int c = 0; c < 4; ++c) {
      v16bf k0 = *reinterpret_cast<const v16bf*>(
          &Kp[(size_t)(kb + lrow) * HD + c * 32 + hi * 16]);
      v16bf k1 = *reinterpret_cast<const v16bf*>(
          &Kp[(size_t)(kb + 16 + lrow) * HD + c * 32 + hi * 16]);
      s0 = wmma_bf16(qf[c], k0, s0);
      s1 = wmma_bf16(qf[c], k1, s1);
    }
    if (kb + 31 > qb0) {
      int col0 = kb + lrow, col1 = col0 + 16;
#pragma unroll
      for (int r = 0; r < 8; ++r) {
        int rowq = qb0 + r + 8 * hi;
        if (col0 > rowq) s0[r] = -1e30f;
        if (col1 > rowq) s1[r] = -1e30f;
      }
    }
    float alpha[8];
#pragma unroll
    for (int r = 0; r < 8; ++r) {
      float mv = fmaxf(s0[r], s1[r]);
#pragma unroll
      for (int off = 8; off > 0; off >>= 1)
        mv = fmaxf(mv, __shfl_xor(mv, off, 32));
      float mn = fmaxf(m[r], mv);
      alpha[r] = __expf(m[r] - mn);
      m[r] = mn;
    }
#pragma unroll
    for (int r = 0; r < 8; ++r) {
      float p0 = __expf(s0[r] - m[r]);
      float p1 = __expf(s1[r] - m[r]);
      float sum = p0 + p1;
#pragma unroll
      for (int off = 8; off > 0; off >>= 1)
        sum += __shfl_xor(sum, off, 32);
      l[r] = l[r] * alpha[r] + sum;
      sP[wave][r + 8 * hi][lrow]      = (bf16_t)p0;
      sP[wave][r + 8 * hi][lrow + 16] = (bf16_t)p1;
#pragma unroll
      for (int nt = 0; nt < 8; ++nt) o[nt][r] *= alpha[r];
    }
    asm volatile("s_wait_dscnt 0" ::: "memory");
    v16bf pf = *reinterpret_cast<const v16bf*>(&sP[wave][lrow][hi * 16]);
#pragma unroll
    for (int nt = 0; nt < 8; ++nt) {
      v16bf vf = *reinterpret_cast<const v16bf*>(
          &Vp[(size_t)(nt * 16 + lrow) * T_SEQ + kb + hi * 16]);
      o[nt] = wmma_bf16(pf, vf, o[nt]);
    }
  }

#pragma unroll
  for (int r = 0; r < 8; ++r) {
    float inv = 1.0f / l[r];
    int q = qb0 + r + 8 * hi;
    size_t base = (size_t)(b * T_SEQ + q) * QSZ + (size_t)h * HD;
#pragma unroll
    for (int nt = 0; nt < 8; ++nt)
      Y[base + nt * 16 + lrow] = (bf16_t)(o[nt][r] * inv);
  }
}

// ---------------------------------------------------------------------------
// Workspace plan (MiB, all stream-ordered so aliasing is safe):
//   [  0, 48) qkv_bf (48)          -> later y_bf (32)
//   [ 48, 80) xb (32)              -> later Qbuf (32)
//   [ 80,128) wqkv_b (48)          -> later Kbuf [80,88) Vbuf [88,96) wo_b [96,128)
// Peak = 128 MiB.
// ---------------------------------------------------------------------------
extern "C" void kernel_launch(void* const* d_in, const int* in_sizes, int n_in,
                              void* d_out, int out_size, void* d_ws,
                              size_t ws_size, hipStream_t stream) {
  (void)in_sizes; (void)n_in; (void)out_size; (void)ws_size;
  const float* x    = (const float*)d_in[0];
  const float* fcos = (const float*)d_in[1];
  const float* fsin = (const float*)d_in[2];
  const float* wqkv = (const float*)d_in[3];
  const float* wo   = (const float*)d_in[4];
  float* out = (float*)d_out;

  const size_t MiB = 1024ull * 1024ull;
  char* ws = (char*)d_ws;
  bf16_t* qkv_bf = (bf16_t*)(ws);
  bf16_t* y_bf   = (bf16_t*)(ws);
  bf16_t* xb     = (bf16_t*)(ws + 48 * MiB);
  bf16_t* Qbuf   = (bf16_t*)(ws + 48 * MiB);
  bf16_t* wqkv_b = (bf16_t*)(ws + 80 * MiB);
  bf16_t* Kbuf   = (bf16_t*)(ws + 80 * MiB);
  bf16_t* Vbuf   = (bf16_t*)(ws + 88 * MiB);
  bf16_t* wo_b   = (bf16_t*)(ws + 96 * MiB);

  // 0) one-time fp32 -> bf16 conversions (bandwidth-bound prep)
  cvt_f32_bf16_kernel<<<4096, 256, 0, stream>>>(x, xb, ROWS * D_MODEL / 4);
  cvt_f32_bf16_kernel<<<6144, 256, 0, stream>>>(wqkv, wqkv_b, QKV_N * D_MODEL / 4);

  // 1) qkv = x @ w_qkv^T  (async-LDS double-buffered WMMA GEMM, bf16 out)
  gemm_wmma_async<false>
      <<<dim3(QKV_N / 128, ROWS / 128), 256, 0, stream>>>(
          xb, wqkv_b, qkv_bf, ROWS, QKV_N, D_MODEL);

  // 2) RoPE + split/transpose (overwrites xb/wqkv_b regions — both dead)
  rope_kernel<<<ROWS, 256, 0, stream>>>(qkv_bf, fcos, fsin, Qbuf, Kbuf, Vbuf);

  // 2b) convert wo (into tail of dead wqkv_b region)
  cvt_f32_bf16_kernel<<<4096, 256, 0, stream>>>(wo, wo_b, D_MODEL * QSZ / 4);

  // 3) causal flash attention (y aliases dead qkv_bf)
  attn_kernel<<<dim3(T_SEQ / 128, NH, BATCH), 256, 0, stream>>>(
      Qbuf, Kbuf, Vbuf, y_bf);

  // 4) out = y @ wo^T  (fp32 out)
  gemm_wmma_async<true>
      <<<dim3(D_MODEL / 128, ROWS / 128), 256, 0, stream>>>(
          y_bf, wo_b, out, ROWS, D_MODEL, QSZ);
}
